// KNNAttention_73547019976907
// MI455X (gfx1250) — matrix operand order repaired and checked by reading, not compiled, not run on hardware
//
#include <hip/hip_runtime.h>
#include <hip/hip_bf16.h>
#include <math.h>

// ---------------------------------------------------------------------------
// CDNA5 (gfx1250) fused gated memory-attention.
// B=2, N=2048, DIM=1024, H=8, DH=64, NUM_BUCKETS=32, MAX_DISTANCE=128
// ---------------------------------------------------------------------------

typedef __attribute__((ext_vector_type(16))) __bf16 v16bf;
typedef __attribute__((ext_vector_type(8)))  float  v8f;

#define B_      2
#define N_      2048
#define DIM_    1024
#define H_      8
#define DH_     64
#define INNER_  512      // H*DH
#define PROJW_  640      // INNER_ + 2*DH (q | k | v per row)
#define MEMPAD_ 2112     // (N+1) padded up to multiple of 64 (33 tiles)

// Async global->LDS DMA (GLOBAL_LOAD_ASYNC_TO_LDS_B128, ASYNCcnt-tracked).
#if defined(__AMDGCN__) && __has_builtin(__builtin_amdgcn_global_load_async_to_lds_b128)
#define HAVE_ASYNC_LDS 1
typedef __attribute__((ext_vector_type(4))) int v4i_async;
typedef __attribute__((address_space(1))) v4i_async* glb_v4i;
typedef __attribute__((address_space(3))) v4i_async* lds_v4i;
#else
#define HAVE_ASYNC_LDS 0
#endif

union BF16Frag { v16bf v; unsigned short u[16]; uint4 q[2]; };

__device__ inline unsigned short f2bf(float f) {
  unsigned int u = __float_as_uint(f);
  unsigned int r = u + 0x7FFFu + ((u >> 16) & 1u);   // round-to-nearest-even
  return (unsigned short)(r >> 16);
}

__device__ inline v8f vzero8() { v8f z = {0,0,0,0,0,0,0,0}; return z; }

__device__ inline v8f wmma_bf16(v16bf a, v16bf b, v8f c) {
  // emits v_wmma_f32_16x16x32_bf16
  return __builtin_amdgcn_wmma_f32_16x16x32_bf16(false, a, false, b,
                                                 (short)0, c, false, false);
}

// Copy 64 bytes (4x b128) from global to LDS for one thread's slice.
__device__ inline void copy64B_to_lds(unsigned short* d, const unsigned short* s) {
#if HAVE_ASYNC_LDS
#pragma unroll
  for (int i = 0; i < 4; ++i)
    __builtin_amdgcn_global_load_async_to_lds_b128(
        (glb_v4i)(s + 8 * i), (lds_v4i)(d + 8 * i), 0, 0);
#else
#pragma unroll
  for (int i = 0; i < 4; ++i) ((uint4*)d)[i] = ((const uint4*)s)[i];
#endif
}

__device__ inline void wait_async_tiles() {
#if HAVE_ASYNC_LDS
#if __has_builtin(__builtin_amdgcn_s_wait_asynccnt)
  __builtin_amdgcn_s_wait_asynccnt(0);
#else
  asm volatile("s_wait_asynccnt 0" ::: "memory");
#endif
#endif
}

// A-matrix fragment (16x32 bf16), memory row-major [rows][stride], base (row0,k0).
// lane<16: row=lane, K = k0+{0..7} then k0+{16..23}
// lane>=16: row=lane-16, K = k0+{8..15} then k0+{24..31}
// stride and k0 must be multiples of 8 halves; base 16B-aligned.
__device__ inline v16bf load_a_frag(const unsigned short* p, int stride, int lane, int k0) {
  int r = lane & 15;
  int koff = k0 + ((lane < 16) ? 0 : 8);
  const unsigned short* row = p + (size_t)r * stride + koff;
  BF16Frag f;
  f.q[0] = *(const uint4*)(row);        // 8 halves
  f.q[1] = *(const uint4*)(row + 16);   // 8 halves
  return f.v;
}

// B-matrix fragment (32x16 bf16) where memory is [N][K] row-major (B = Mem^T).
// element (K=k, N=n) = mem[n*stride + k]; K order matches load_a_frag (k0+8h+{0..7}, k0+16+8h+{0..7}).
__device__ inline v16bf load_b_fragT(const unsigned short* mem, int stride,
                                     int n0, int k0, int lane) {
  int n = n0 + (lane & 15);
  int koff = k0 + ((lane < 16) ? 0 : 8);
  const unsigned short* p = mem + (size_t)n * stride + koff;
  BF16Frag f;
  f.q[0] = *(const uint4*)(p);
  f.q[1] = *(const uint4*)(p + 16);
  return f.v;
}

// T5 relative-position bias: dist = i - j >= 0 (causal side).
__device__ inline float rel_bias(int dist, const float* rel_emb, int h) {
  int bucket;
  if (dist < 16) {
    bucket = dist;
  } else {
    // 16 + log(dist/16)/log(8) * 16, truncated, capped at 31
    float t = __logf((float)dist * (1.0f / 16.0f)) * (16.0f / 2.0794415416798357f);
    bucket = 16 + (int)t;
    if (bucket > 31) bucket = 31;
  }
  return rel_emb[bucket * H_ + h] * 8.0f; // * sqrt(DH)
}

// Cooperative 64x64 bf16 tile load (128 threads), async DMA when available.
__device__ inline void load_tile64(unsigned short* dst, const unsigned short* src, int stride) {
  int t  = threadIdx.x;        // 0..127
  int r  = t >> 1;             // 0..63
  int c0 = (t & 1) * 32;
  copy64B_to_lds(dst + r * 64 + c0, src + (size_t)r * stride + c0);
}

// ---------------------------------------------------------------------------
// Kernel 1: pack mem_kv (B, H*N, 2, DH) f32 -> bf16
//   memK  [B][H][2112][64]  (row-major keys)
//   memVt [B][H][64][2112]  (transposed values: dh-major, for B fragments)
// row/col 0 = null_k/null_v, 1..N = mem rows, N+1.. = zero pad.
// ---------------------------------------------------------------------------
__global__ __launch_bounds__(256)
void pack_mem_kernel(const float* __restrict__ mem_kv,
                     const float* __restrict__ null_k,
                     const float* __restrict__ null_v,
                     unsigned short* __restrict__ memK,
                     unsigned short* __restrict__ memVt) {
  size_t idx = (size_t)blockIdx.x * 256 + threadIdx.x;
  const size_t total = (size_t)B_ * H_ * MEMPAD_ * DH_;
  if (idx >= total) return;
  int dh = (int)(idx & (DH_ - 1));
  size_t r = idx >> 6;
  int j = (int)(r % MEMPAD_);
  size_t bh = r / MEMPAD_;                 // b*H + h
  float vk, vv;
  if (j == 0) {
    vk = null_k[dh]; vv = null_v[dh];
  } else if (j <= N_) {
    size_t base = ((bh * (size_t)N_ + (size_t)(j - 1)) * 2) * DH_ + dh;
    vk = mem_kv[base];
    vv = mem_kv[base + DH_];
  } else {
    vk = 0.0f; vv = 0.0f;
  }
  memK[idx] = f2bf(vk);
  memVt[(bh * DH_ + (size_t)dh) * MEMPAD_ + j] = f2bf(vv);
}

// ---------------------------------------------------------------------------
// Kernel 2: fused QKV projection: (4096 x 1024) @ (1024 x 640) -> bf16 proj.
// Block tile 128x128, 8 waves (4x2), LDS-staged, bf16 WMMA.
// B tile staged TRANSPOSED ([N][K]) so B fragments are contiguous b128 loads.
// Also scatters V columns into vT [B][DH][N].
// ---------------------------------------------------------------------------
__global__ __launch_bounds__(256)
void qkv_proj_kernel(const float* __restrict__ x,
                     const float* __restrict__ Wq,
                     const float* __restrict__ Wkv,
                     unsigned short* __restrict__ proj,
                     unsigned short* __restrict__ vT) {
  __shared__ unsigned short ldsA[128 * 64];    // [M][K]
  __shared__ unsigned short ldsBT[128 * 64];   // [N][K] (transposed)
  const int m0 = blockIdx.x * 128;
  const int n0 = blockIdx.y * 128;
  const int lane = threadIdx.x & 31;
  const int wave = threadIdx.x >> 5;
  const int wm = wave & 3;       // 4 waves along M (32 rows each)
  const int wn = wave >> 2;      // 2 waves along N (64 cols each)
  const int hi = lane >> 4, ln = lane & 15;

  const float* Bsrc; int bstride, bcol;
  if (n0 < INNER_) { Bsrc = Wq;  bstride = INNER_;  bcol = n0; }
  else             { Bsrc = Wkv; bstride = 2 * DH_; bcol = n0 - INNER_; }

  v8f acc[2][4];
#pragma unroll
  for (int i = 0; i < 2; ++i)
#pragma unroll
    for (int j = 0; j < 4; ++j) acc[i][j] = vzero8();

  for (int k0 = 0; k0 < DIM_; k0 += 64) {
    __syncthreads();
    { // A tile: 128x64 f32 -> bf16 LDS [M][K]
      int t = threadIdx.x; int r = t >> 1; int c0 = (t & 1) * 32;
      const float* s = x + (size_t)(m0 + r) * DIM_ + k0 + c0;
      unsigned short* d = ldsA + r * 64 + c0;
#pragma unroll
      for (int i = 0; i < 32; ++i) d[i] = f2bf(s[i]);
    }
    { // B tile transposed: LDS[n][k] = Bsrc[(k0+k)*bstride + bcol + n]
      int t = threadIdx.x; int n = t >> 1; int kk0 = (t & 1) * 32;
      unsigned short* d = ldsBT + n * 64 + kk0;
#pragma unroll
      for (int i = 0; i < 32; ++i)
        d[i] = f2bf(Bsrc[(size_t)(k0 + kk0 + i) * bstride + bcol + n]);
    }
    __syncthreads();
#pragma unroll
    for (int kk = 0; kk < 64; kk += 32) {
      v16bf af[2], bfr[4];
#pragma unroll
      for (int i = 0; i < 2; ++i)
        af[i] = load_a_frag(ldsA + (wm * 32 + i * 16) * 64, 64, lane, kk);
#pragma unroll
      for (int j = 0; j < 4; ++j)
        bfr[j] = load_b_fragT(ldsBT, 64, wn * 64 + j * 16, kk, lane);
#pragma unroll
      for (int i = 0; i < 2; ++i)
#pragma unroll
        for (int j = 0; j < 4; ++j) acc[i][j] = wmma_bf16(af[i], bfr[j], acc[i][j]);
    }
  }
#pragma unroll
  for (int i = 0; i < 2; ++i)
#pragma unroll
    for (int j = 0; j < 4; ++j)
#pragma unroll
      for (int e = 0; e < 8; ++e) {
        int r = m0 + wm * 32 + i * 16 + e + 8 * hi;
        int c = n0 + wn * 64 + j * 16 + ln;
        unsigned short bfv = f2bf(acc[i][j][e]);
        proj[(size_t)r * PROJW_ + c] = bfv;
        if (c >= INNER_ + DH_) {               // v columns -> transposed buffer
          int bb = r >> 11;                    // r / N_
          int nn = r & (N_ - 1);
          vT[((size_t)bb * DH_ + (c - (INNER_ + DH_))) * N_ + nn] = bfv;
        }
      }
}

// ---------------------------------------------------------------------------
// Kernel 3: fused flash attention, both branches + gated blend.
// Grid (N/64, H, B), 128 threads = 4 waves; each wave owns 16 query rows.
// K/V tiles streamed to LDS with async DMA; P staged per-wave for P*V WMMA.
// ---------------------------------------------------------------------------
__global__ __launch_bounds__(128)
void attn_fused_kernel(const unsigned short* __restrict__ proj,
                       const unsigned short* __restrict__ vT,
                       const unsigned short* __restrict__ memK,
                       const unsigned short* __restrict__ memVt,
                       const unsigned char*  __restrict__ mem_mask,
                       const float* __restrict__ rel_emb,
                       const float* __restrict__ gate,
                       unsigned short* __restrict__ comb) {
  __shared__ unsigned short ldsK[64 * 64];       // [key][dh]
  __shared__ unsigned short ldsVt[64 * 64];      // [dh][key]  (transposed)
  __shared__ unsigned short ldsP[4][16 * 64];    // per-wave P [row][key]

  const int tile = blockIdx.x;
  const int h    = blockIdx.y;
  const int b    = blockIdx.z;
  const int lane = threadIdx.x & 31;
  const int wave = threadIdx.x >> 5;
  const int hi = lane >> 4, ln = lane & 15;
  const int m_base = tile * 64 + wave * 16;
  const int tr = threadIdx.x >> 1;               // 0..63 tile row for prefetch

  // Q fragments for this wave's 16 rows (bf16, dh split in two K=32 chunks)
  const unsigned short* qbase = proj + ((size_t)(b * N_ + m_base)) * PROJW_ + h * DH_;
  v16bf qf0 = load_a_frag(qbase, PROJW_, lane, 0);
  v16bf qf1 = load_a_frag(qbase, PROJW_, lane, 32);

  const unsigned short* kS = proj + (size_t)b * N_ * PROJW_ + INNER_;  // k rows
  const unsigned short* vS = vT + (size_t)b * DH_ * N_;                // vT rows

  float m_r[8], l_r[8];
  v8f O[4];
#pragma unroll
  for (int e = 0; e < 8; ++e) { m_r[e] = -1e30f; l_r[e] = 0.0f; }
#pragma unroll
  for (int t = 0; t < 4; ++t) O[t] = vzero8();

  // ------------------- local causal branch -------------------
  for (int kt = 0; kt <= tile; ++kt) {
    __syncthreads();
    load_tile64(ldsK,  kS + (size_t)(kt * 64) * PROJW_, PROJW_);
    load_tile64(ldsVt, vS + (kt * 64), N_);     // rows=dh, cols=keys
    if (kt + 1 <= tile) {                       // prefetch next tiles into cache
      __builtin_prefetch(kS + (size_t)((kt + 1) * 64 + tr) * PROJW_, 0, 0);
      __builtin_prefetch(vS + (size_t)tr * N_ + (kt + 1) * 64, 0, 0);
    }
    wait_async_tiles();
    __syncthreads();

    float sv[4][8];
#pragma unroll
    for (int ct = 0; ct < 4; ++ct) {
      v8f S = vzero8();
      S = wmma_bf16(qf0, load_b_fragT(ldsK, 64, ct * 16, 0,  lane), S);
      S = wmma_bf16(qf1, load_b_fragT(ldsK, 64, ct * 16, 32, lane), S);
      int jg = kt * 64 + ct * 16 + ln;
#pragma unroll
      for (int e = 0; e < 8; ++e) {
        int ig = m_base + e + 8 * hi;
        float s = S[e] * 0.125f;                       // 1/sqrt(DH)
        sv[ct][e] = (jg > ig) ? -3.0e38f : (s + rel_bias(ig - jg, rel_emb, h));
      }
    }
    float alpha[8];
#pragma unroll
    for (int e = 0; e < 8; ++e) {
      float mx = fmaxf(fmaxf(sv[0][e], sv[1][e]), fmaxf(sv[2][e], sv[3][e]));
#pragma unroll
      for (int off = 8; off >= 1; off >>= 1) mx = fmaxf(mx, __shfl_xor(mx, off, 32));
      float mnew = fmaxf(m_r[e], mx);
      alpha[e] = __expf(m_r[e] - mnew);
      m_r[e] = mnew;
      float rs = 0.0f;
#pragma unroll
      for (int ct = 0; ct < 4; ++ct) {
        float p = __expf(sv[ct][e] - mnew);
        sv[ct][e] = p; rs += p;
      }
#pragma unroll
      for (int off = 8; off >= 1; off >>= 1) rs += __shfl_xor(rs, off, 32);
      l_r[e] = l_r[e] * alpha[e] + rs;
    }
#pragma unroll
    for (int t = 0; t < 4; ++t)
#pragma unroll
      for (int e = 0; e < 8; ++e) O[t][e] *= alpha[e];
#pragma unroll
    for (int ct = 0; ct < 4; ++ct)
#pragma unroll
      for (int e = 0; e < 8; ++e)
        ldsP[wave][(e + 8 * hi) * 64 + ct * 16 + ln] = f2bf(sv[ct][e]);
    asm volatile("s_wait_dscnt 0" ::: "memory");   // P write -> A-frag read
#pragma unroll
    for (int nt = 0; nt < 4; ++nt)
#pragma unroll
      for (int kc = 0; kc < 64; kc += 32) {
        v16bf ap = load_a_frag(ldsP[wave], 64, lane, kc);
        v16bf bv = load_b_fragT(ldsVt, 64, nt * 16, kc, lane);
        O[nt] = wmma_bf16(ap, bv, O[nt]);
      }
  }
  float lo[4][8];
#pragma unroll
  for (int t = 0; t < 4; ++t)
#pragma unroll
    for (int e = 0; e < 8; ++e) lo[t][e] = O[t][e] / l_r[e];

  // ------------------- memory branch (non-causal, N+1 keys, padded) -------------------
#pragma unroll
  for (int e = 0; e < 8; ++e) { m_r[e] = -1e30f; l_r[e] = 0.0f; }
#pragma unroll
  for (int t = 0; t < 4; ++t) O[t] = vzero8();

  const unsigned short* mkb = memK  + ((size_t)(b * H_ + h)) * MEMPAD_ * DH_;   // [j][dh]
  const unsigned short* mvb = memVt + ((size_t)(b * H_ + h)) * DH_ * MEMPAD_;   // [dh][j]
  const unsigned char*  mmb = mem_mask + (size_t)b * (H_ * N_) + (size_t)h * N_;

  for (int kt = 0; kt < MEMPAD_ / 64; ++kt) {
    __syncthreads();
    load_tile64(ldsK,  mkb + (size_t)(kt * 64) * DH_, DH_);
    load_tile64(ldsVt, mvb + (kt * 64), MEMPAD_);
    if (kt + 1 < MEMPAD_ / 64) {
      __builtin_prefetch(mkb + (size_t)((kt + 1) * 64 + tr) * DH_, 0, 0);
      __builtin_prefetch(mvb + (size_t)tr * MEMPAD_ + (kt + 1) * 64, 0, 0);
    }
    wait_async_tiles();
    __syncthreads();

    float sv[4][8];
#pragma unroll
    for (int ct = 0; ct < 4; ++ct) {
      v8f S = vzero8();
      S = wmma_bf16(qf0, load_b_fragT(ldsK, 64, ct * 16, 0,  lane), S);
      S = wmma_bf16(qf1, load_b_fragT(ldsK, 64, ct * 16, 32, lane), S);
      int jg = kt * 64 + ct * 16 + ln;
      bool valid = (jg == 0) || ((jg <= N_) && (mmb[jg - 1] != 0));
#pragma unroll
      for (int e = 0; e < 8; ++e)
        sv[ct][e] = valid ? (S[e] * 0.125f) : -3.0e38f;
    }
    float alpha[8];
#pragma unroll
    for (int e = 0; e < 8; ++e) {
      float mx = fmaxf(fmaxf(sv[0][e], sv[1][e]), fmaxf(sv[2][e], sv[3][e]));
#pragma unroll
      for (int off = 8; off >= 1; off >>= 1) mx = fmaxf(mx, __shfl_xor(mx, off, 32));
      float mnew = fmaxf(m_r[e], mx);
      alpha[e] = __expf(m_r[e] - mnew);
      m_r[e] = mnew;
      float rs = 0.0f;
#pragma unroll
      for (int ct = 0; ct < 4; ++ct) {
        float p = __expf(sv[ct][e] - mnew);
        sv[ct][e] = p; rs += p;
      }
#pragma unroll
      for (int off = 8; off >= 1; off >>= 1) rs += __shfl_xor(rs, off, 32);
      l_r[e] = l_r[e] * alpha[e] + rs;
    }
#pragma unroll
    for (int t = 0; t < 4; ++t)
#pragma unroll
      for (int e = 0; e < 8; ++e) O[t][e] *= alpha[e];
#pragma unroll
    for (int ct = 0; ct < 4; ++ct)
#pragma unroll
      for (int e = 0; e < 8; ++e)
        ldsP[wave][(e + 8 * hi) * 64 + ct * 16 + ln] = f2bf(sv[ct][e]);
    asm volatile("s_wait_dscnt 0" ::: "memory");
#pragma unroll
    for (int nt = 0; nt < 4; ++nt)
#pragma unroll
      for (int kc = 0; kc < 64; kc += 32) {
        v16bf ap = load_a_frag(ldsP[wave], 64, lane, kc);
        v16bf bv = load_b_fragT(ldsVt, 64, nt * 16, kc, lane);
        O[nt] = wmma_bf16(ap, bv, O[nt]);
      }
  }

  // ------------------- gated blend, write combined bf16 [B*N, 512] -------------------
  float g = 1.0f / (1.0f + __expf(-gate[h]));
#pragma unroll
  for (int t = 0; t < 4; ++t)
#pragma unroll
    for (int e = 0; e < 8; ++e) {
      float mo  = O[t][e] / l_r[e];
      float val = lo[t][e] * g + mo * (1.0f - g);
      int ig = m_base + e + 8 * hi;
      comb[((size_t)(b * N_ + ig)) * INNER_ + h * DH_ + t * 16 + ln] = f2bf(val);
    }
}

// ---------------------------------------------------------------------------
// Kernel 4: output projection (4096 x 512) @ (512 x 1024) + bout -> f32 out.
// A tile async-DMA'd (already bf16); B tile staged transposed.
// ---------------------------------------------------------------------------
__global__ __launch_bounds__(256)
void out_proj_kernel(const unsigned short* __restrict__ comb,
                     const float* __restrict__ Wout,
                     const float* __restrict__ bout,
                     float* __restrict__ out) {
  __shared__ unsigned short ldsA[128 * 64];    // [M][K]
  __shared__ unsigned short ldsBT[128 * 64];   // [N][K]
  const int m0 = blockIdx.x * 128;
  const int n0 = blockIdx.y * 128;
  const int lane = threadIdx.x & 31;
  const int wave = threadIdx.x >> 5;
  const int wm = wave & 3;
  const int wn = wave >> 2;
  const int hi = lane >> 4, ln = lane & 15;

  v8f acc[2][4];
#pragma unroll
  for (int i = 0; i < 2; ++i)
#pragma unroll
    for (int j = 0; j < 4; ++j) acc[i][j] = vzero8();

  for (int k0 = 0; k0 < INNER_; k0 += 64) {
    __syncthreads();
    { // A tile: 128x64 bf16 -> LDS via async DMA (256 threads, 32 halves each)
      int t = threadIdx.x; int r = t >> 1; int c0 = (t & 1) * 32;
      copy64B_to_lds(ldsA + r * 64 + c0,
                     comb + (size_t)(m0 + r) * INNER_ + k0 + c0);
    }
    { // B tile transposed: LDS[n][k] = Wout[(k0+k)*DIM_ + n0 + n]
      int t = threadIdx.x; int n = t >> 1; int kk0 = (t & 1) * 32;
      unsigned short* d = ldsBT + n * 64 + kk0;
#pragma unroll
      for (int i = 0; i < 32; ++i)
        d[i] = f2bf(Wout[(size_t)(k0 + kk0 + i) * DIM_ + n0 + n]);
    }
    wait_async_tiles();
    __syncthreads();
#pragma unroll
    for (int kk = 0; kk < 64; kk += 32) {
      v16bf af[2], bfr[4];
#pragma unroll
      for (int i = 0; i < 2; ++i)
        af[i] = load_a_frag(ldsA + (wm * 32 + i * 16) * 64, 64, lane, kk);
#pragma unroll
      for (int j = 0; j < 4; ++j)
        bfr[j] = load_b_fragT(ldsBT, 64, wn * 64 + j * 16, kk, lane);
#pragma unroll
      for (int i = 0; i < 2; ++i)
#pragma unroll
        for (int j = 0; j < 4; ++j) acc[i][j] = wmma_bf16(af[i], bfr[j], acc[i][j]);
    }
  }
#pragma unroll
  for (int i = 0; i < 2; ++i)
#pragma unroll
    for (int j = 0; j < 4; ++j)
#pragma unroll
      for (int e = 0; e < 8; ++e) {
        int r = m0 + wm * 32 + i * 16 + e + 8 * hi;
        int c = n0 + wn * 64 + j * 16 + ln;
        out[(size_t)r * DIM_ + c] = acc[i][j][e] + bout[c];
      }
}

// ---------------------------------------------------------------------------
// Host launcher
// ---------------------------------------------------------------------------
extern "C" void kernel_launch(void* const* d_in, const int* in_sizes, int n_in,
                              void* d_out, int out_size, void* d_ws, size_t ws_size,
                              hipStream_t stream) {
  const float* x        = (const float*)d_in[0];
  const float* mem_kv   = (const float*)d_in[1];
  const unsigned char* mem_mask = (const unsigned char*)d_in[2];
  const float* Wq       = (const float*)d_in[3];
  const float* Wkv      = (const float*)d_in[4];
  const float* Wout     = (const float*)d_in[5];
  const float* bout     = (const float*)d_in[6];
  const float* rel_emb  = (const float*)d_in[7];
  const float* gate     = (const float*)d_in[8];
  const float* null_k   = (const float*)d_in[9];
  const float* null_v   = (const float*)d_in[10];

  (void)in_sizes; (void)n_in; (void)out_size; (void)ws_size;

  // Workspace layout (bf16):
  //   proj  [4096][640]          5,242,880 B
  //   memK  [2][8][2112][64]     4,325,376 B
  //   memVt [2][8][64][2112]     4,325,376 B
  //   comb  [4096][512]          4,194,304 B
  //   vT    [2][64][2048]          524,288 B
  char* ws = (char*)d_ws;
  const size_t projBytes = (size_t)(B_ * N_) * PROJW_ * 2;
  const size_t memBytes  = (size_t)B_ * H_ * MEMPAD_ * DH_ * 2;
  const size_t combBytes = (size_t)(B_ * N_) * INNER_ * 2;
  unsigned short* proj  = (unsigned short*)(ws);
  unsigned short* memK  = (unsigned short*)(ws + projBytes);
  unsigned short* memVt = (unsigned short*)(ws + projBytes + memBytes);
  unsigned short* comb  = (unsigned short*)(ws + projBytes + 2 * memBytes);
  unsigned short* vT    = (unsigned short*)(ws + projBytes + 2 * memBytes + combBytes);

  const size_t packElems = (size_t)B_ * H_ * MEMPAD_ * DH_;
  pack_mem_kernel<<<dim3((unsigned)((packElems + 255) / 256)), 256, 0, stream>>>(
      mem_kv, null_k, null_v, memK, memVt);

  qkv_proj_kernel<<<dim3((B_ * N_) / 128, PROJW_ / 128), 256, 0, stream>>>(
      x, Wq, Wkv, proj, vT);

  attn_fused_kernel<<<dim3(N_ / 64, H_, B_), 128, 0, stream>>>(
      proj, vT, memK, memVt, mem_mask, rel_emb, gate, comb);

  out_proj_kernel<<<dim3((B_ * N_) / 128, DIM_ / 128), 256, 0, stream>>>(
      comb, Wout, bout, (float*)d_out);
}
